// Quantizer1d_15547781611764
// MI455X (gfx1250) — compile-verified, matrix operand order checked by reading, not run on hardware
//
#include <hip/hip_runtime.h>

typedef __attribute__((ext_vector_type(16))) _Float16     v16h;
typedef __attribute__((ext_vector_type(8)))  float        v8f;
typedef __attribute__((ext_vector_type(4)))  unsigned int v4u;
typedef __attribute__((ext_vector_type(8)))  int          v8i;
typedef __attribute__((ext_vector_type(4)))  int          v4i;

#define C_DIM   512
#define K_DIM   1024
#define T_DIM   2048
#define B_DIM   16
#define M_TILE  256                       // rows per workgroup (8 waves x 32 rows)
#define NWG     (B_DIM * T_DIM / M_TILE)  // 128 workgroups
#define CHUNK_K 64                        // emb rows staged per phase
#define NPHASE  (K_DIM / CHUNK_K)         // 16 phases

union Frag16 { v16h v; v4u u[2]; };

// ---------------- Kernel 1: emb -> f16, e2 = ||e_k||^2 (f32) ----------------
__global__ __launch_bounds__(128) void prep_emb(const float* __restrict__ emb,
                                                _Float16* __restrict__ embh,
                                                float* __restrict__ e2) {
    const int k = blockIdx.x;
    const float* row = emb + (size_t)k * C_DIM;
    _Float16* orow = embh + (size_t)k * C_DIM;
    float s = 0.f;
    for (int c = threadIdx.x; c < C_DIM; c += 128) {
        float v = row[c];
        s = __builtin_fmaf(v, v, s);
        orow[c] = (_Float16)v;
    }
    for (int off = 16; off; off >>= 1) s += __shfl_xor(s, off, 32);
    __shared__ float red[4];
    const int wid = threadIdx.x >> 5, lane = threadIdx.x & 31;
    if (lane == 0) red[wid] = s;
    __syncthreads();
    if (threadIdx.x == 0) e2[k] = red[0] + red[1] + red[2] + red[3];
}

// Convert 16 f32 (two 8-element chunks per the A/B fragment layout) -> v16h
__device__ __forceinline__ v16h cvt_frag(const float* __restrict__ p) {
    float4 f0 = *(const float4*)(p);
    float4 f1 = *(const float4*)(p + 4);
    float4 f2 = *(const float4*)(p + 16);
    float4 f3 = *(const float4*)(p + 20);
    v16h r;
    r[0]  = (_Float16)f0.x; r[1]  = (_Float16)f0.y; r[2]  = (_Float16)f0.z; r[3]  = (_Float16)f0.w;
    r[4]  = (_Float16)f1.x; r[5]  = (_Float16)f1.y; r[6]  = (_Float16)f1.z; r[7]  = (_Float16)f1.w;
    r[8]  = (_Float16)f2.x; r[9]  = (_Float16)f2.y; r[10] = (_Float16)f2.z; r[11] = (_Float16)f2.w;
    r[12] = (_Float16)f3.x; r[13] = (_Float16)f3.y; r[14] = (_Float16)f3.z; r[15] = (_Float16)f3.w;
    return r;
}

#if __has_builtin(__builtin_amdgcn_tensor_load_to_lds)
#define HAVE_TDM 1
#endif

// Stage CHUNK_K rows x C_DIM f16 (row-major, stride C_DIM) from global into LDS.
__device__ __forceinline__ void tdm_stage(const _Float16* __restrict__ gsrc,
                                          unsigned lds_off) {
#ifdef HAVE_TDM
    unsigned long long ga = (unsigned long long)(uintptr_t)gsrc;
    v4u g0;
    g0[0] = 1u;                                           // count=1, user descriptor
    g0[1] = lds_off;                                      // lds_addr (bytes)
    g0[2] = (unsigned)(ga & 0xffffffffull);               // global_addr lo
    g0[3] = (unsigned)((ga >> 32) & 0x01ffffffu) | (2u << 30); // addr hi | type=2
    v8i g1;
    g1[0] = (int)(1u << 16);                              // data_size = 2 bytes
    g1[1] = (int)((unsigned)(C_DIM & 0xffff) << 16);      // tensor_dim0 lo16
    g1[2] = (int)((unsigned)K_DIM << 16);                 // tensor_dim0 hi | tensor_dim1 lo
    g1[3] = (int)((unsigned)(C_DIM & 0xffff) << 16);      // tensor_dim1 hi | tile_dim0
    g1[4] = CHUNK_K;                                      // tile_dim1 | tile_dim2=0
    g1[5] = C_DIM;                                        // tensor_dim0_stride lo32
    g1[6] = 0;
    g1[7] = 0;
    v4i g2 = {0, 0, 0, 0};
    v4i g3 = {0, 0, 0, 0};
    v8i g4 = {0, 0, 0, 0, 0, 0, 0, 0};                    // 6-arg toolchain: extra group
    __builtin_amdgcn_tensor_load_to_lds(g0, g1, g2, g3, g4, 0);
#endif
}

__device__ __forceinline__ void tdm_wait0() {
#ifdef HAVE_TDM
    __builtin_amdgcn_s_wait_tensorcnt(0);
#endif
}

// ---------------- Kernel 2: main VQ ----------------
__global__ __launch_bounds__(256) void vq_main(const float* __restrict__ x,
                                               const float* __restrict__ emb,
                                               const _Float16* __restrict__ embh,
                                               const float* __restrict__ e2,
                                               float* __restrict__ out_q,
                                               float* __restrict__ out_idx,
                                               float* __restrict__ partial) {
    __shared__ __align__(16) _Float16 stage[2][CHUNK_K * C_DIM];  // 2 x 64 KB
    __shared__ int   sfin[M_TILE];
    __shared__ float sloss[8];

    const int tid  = threadIdx.x;
    const int wg   = blockIdx.x;
    const int m0   = wg * M_TILE;
    const int wid  = tid >> 5;
    const int lane = tid & 31;
    const int colL = lane & 15;
    const int hL   = lane >> 4;
    const int wrow = m0 + wid * 32;          // this wave's 32 rows

    // Kick off staging of chunk 0 ASAP (overlaps with A-fragment build)
    if (wid == 0)
        tdm_stage(embh, (unsigned)(uintptr_t)&stage[0][0]);
#ifndef HAVE_TDM
    // fallback: cooperative copy of chunk 0
    for (int i = tid * 8; i < CHUNK_K * C_DIM; i += 256 * 8)
        *(v4u*)&stage[0][i] = *(const v4u*)&embh[i];
#endif

    // ---- Register-resident A: 2 sets of 16 rows, full K, f32->f16 on the fly ----
    v16h a0[16], a1[16];
    {
        const float* r0 = x + (size_t)(wrow + colL) * C_DIM;
        const float* r1 = x + (size_t)(wrow + 16 + colL) * C_DIM;
        #pragma unroll
        for (int s = 0; s < 16; ++s) {
            const int c0 = 32 * s + 8 * hL;
            a0[s] = cvt_frag(r0 + c0);
            a1[s] = cvt_frag(r1 + c0);
        }
    }

    float best0[8], best1[8];
    int   bidx0[8], bidx1[8];
    #pragma unroll
    for (int j = 0; j < 8; ++j) { best0[j] = best1[j] = 3.4e38f; bidx0[j] = bidx1[j] = 0; }

    // ---- Double-buffered sweep over 16 chunks of 64 codes ----
    for (int p = 0; p < NPHASE; ++p) {
        if (wid == 0) tdm_wait0();          // chunk p landed
        __syncthreads();                    // release buf[p&1]; all done reading buf[(p+1)&1]
        if (wid == 0 && p + 1 < NPHASE)
            tdm_stage(embh + (size_t)(p + 1) * CHUNK_K * C_DIM,
                      (unsigned)(uintptr_t)&stage[(p + 1) & 1][0]);
#ifndef HAVE_TDM
        if (p + 1 < NPHASE) {
            // fallback path would copy synchronously here
        }
#endif
        const _Float16* sb = &stage[p & 1][0];

        for (int tb = 0; tb < 2; ++tb) {    // two 32-code register tiles per chunk
            const int kb = p * CHUNK_K + tb * 32;
            const _Float16* br0 = sb + (size_t)(tb * 32 + colL) * C_DIM;
            const _Float16* br1 = sb + (size_t)(tb * 32 + 16 + colL) * C_DIM;
            v8f acc00 = {}, acc01 = {}, acc10 = {}, acc11 = {};
            #pragma unroll
            for (int s = 0; s < 16; ++s) {
                const int c0 = 32 * s + 8 * hL;
                Frag16 fb0, fb1;
                fb0.u[0] = *(const v4u*)(br0 + c0);
                fb0.u[1] = *(const v4u*)(br0 + c0 + 16);
                fb1.u[0] = *(const v4u*)(br1 + c0);
                fb1.u[1] = *(const v4u*)(br1 + c0 + 16);
                acc00 = __builtin_amdgcn_wmma_f32_16x16x32_f16(false, a0[s], false, fb0.v, (short)0, acc00, false, false);
                acc10 = __builtin_amdgcn_wmma_f32_16x16x32_f16(false, a1[s], false, fb0.v, (short)0, acc10, false, false);
                acc01 = __builtin_amdgcn_wmma_f32_16x16x32_f16(false, a0[s], false, fb1.v, (short)0, acc01, false, false);
                acc11 = __builtin_amdgcn_wmma_f32_16x16x32_f16(false, a1[s], false, fb1.v, (short)0, acc11, false, false);
            }
            const float e2a = e2[kb + colL];
            const float e2b = e2[kb + 16 + colL];
            const int   kia = kb + colL, kib = kb + 16 + colL;
            #pragma unroll
            for (int j = 0; j < 8; ++j) {
                float d;
                d = __builtin_fmaf(-2.f, acc00[j], e2a); if (d < best0[j]) { best0[j] = d; bidx0[j] = kia; }
                d = __builtin_fmaf(-2.f, acc01[j], e2b); if (d < best0[j]) { best0[j] = d; bidx0[j] = kib; }
                d = __builtin_fmaf(-2.f, acc10[j], e2a); if (d < best1[j]) { best1[j] = d; bidx1[j] = kia; }
                d = __builtin_fmaf(-2.f, acc11[j], e2b); if (d < best1[j]) { best1[j] = d; bidx1[j] = kib; }
            }
        }
    }

    // ---- Cross-lane argmin within 16-lane halves (rows j / j+8), per A set ----
    #pragma unroll
    for (int set = 0; set < 2; ++set) {
        float* bb = set ? best1 : best0;
        int*   ii = set ? bidx1 : bidx0;
        #pragma unroll
        for (int j = 0; j < 8; ++j) {
            float b = bb[j]; int bi = ii[j];
            #pragma unroll
            for (int off = 8; off >= 1; off >>= 1) {
                float ob = __shfl_xor(b, off, 32);
                int   oi = __shfl_xor(bi, off, 32);
                if (ob < b || (ob == b && oi < bi)) { b = ob; bi = oi; }
            }
            if (colL == 0) {
                const int lrow = wid * 32 + set * 16 + j + hL * 8;
                sfin[lrow] = bi;
                out_idx[m0 + lrow] = (float)bi;
            }
        }
    }
    __syncthreads();

    // ---- Phase C: gather quant (STE arithmetic) + normalized-MSE loss (f32) ----
    float lsum = 0.f;
    for (int rr = 0; rr < 32; ++rr) {
        const int lrow = wid * 32 + rr;
        const int fin  = sfin[lrow];
        const float* xr = x     + (size_t)(m0 + lrow) * C_DIM;
        const float* qr = emb   + (size_t)fin         * C_DIM;
        float* orow     = out_q + (size_t)(m0 + lrow) * C_DIM;
        float dot = 0.f, sx2 = 0.f;
        for (int c = lane; c < C_DIM; c += 32) {
            float xv = xr[c], qv = qr[c];
            dot = __builtin_fmaf(xv, qv, dot);
            sx2 = __builtin_fmaf(xv, xv, sx2);
            orow[c] = xv + (qv - xv);             // match reference STE rounding
        }
        for (int off = 16; off; off >>= 1) {
            dot += __shfl_xor(dot, off, 32);
            sx2 += __shfl_xor(sx2, off, 32);
        }
        if (lane == 0) {
            float sq2 = e2[fin];
            float dx = fmaxf(sqrtf(sx2), 1e-5f);
            float dq = fmaxf(sqrtf(sq2), 1e-5f);
            lsum += sx2 / (dx * dx) + sq2 / (dq * dq) - 2.f * dot / (dx * dq);
        }
    }
    if (lane == 0) sloss[wid] = lsum;
    __syncthreads();
    if (tid == 0) {   // deterministic sequential combine of 8 wave partials
        float s = 0.f;
        for (int w = 0; w < 8; ++w) s += sloss[w];
        partial[wg] = s;
    }
}

// ---------------- Kernel 3: finalize per-batch losses ----------------
__global__ void finalize(const float* __restrict__ partial,
                         float* __restrict__ out_cb,
                         float* __restrict__ out_cm) {
    const int b = threadIdx.x;
    if (b < B_DIM) {
        float s = 0.f;
        const int wpb = NWG / B_DIM;   // 8 WGs per batch
        for (int i = 0; i < wpb; ++i) s += partial[b * wpb + i];
        float loss = s / (float)(T_DIM * C_DIM);
        out_cb[b] = loss;
        out_cm[b] = loss;
    }
}

extern "C" void kernel_launch(void* const* d_in, const int* in_sizes, int n_in,
                              void* d_out, int out_size, void* d_ws, size_t ws_size,
                              hipStream_t stream) {
    const float* x   = (const float*)d_in[0];   // [16,2048,512] f32
    const float* emb = (const float*)d_in[1];   // [1024,512]    f32

    float* out     = (float*)d_out;
    float* out_q   = out;                                        // [B,T,C]
    float* out_cb  = out + (size_t)B_DIM * T_DIM * C_DIM;        // [B]
    float* out_cm  = out_cb + B_DIM;                             // [B]
    float* out_idx = out_cm + B_DIM;                             // [B,T] as float

    char* ws = (char*)d_ws;
    _Float16* embh   = (_Float16*)ws;                                   // 1 MB
    float*    e2     = (float*)(ws + (size_t)K_DIM * C_DIM * 2);        // 4 KB
    float*    partic = (float*)(ws + (size_t)K_DIM * C_DIM * 2 + 4096); // 512 B

    prep_emb<<<K_DIM, 128, 0, stream>>>(emb, embh, e2);
    vq_main <<<NWG, 256, 0, stream>>>(x, emb, embh, e2, out_q, out_idx, partic);
    finalize<<<1, 32, 0, stream>>>(partic, out_cb, out_cm);
}
